// DAMPLayer_4930622456346
// MI455X (gfx1250) — compile-verified, matrix operand order checked by reading, not run on hardware
//
#include <hip/hip_runtime.h>
#include <hip/hip_fp16.h>
#include <math.h>

// ---------- CDNA5 WMMA types ----------
typedef __attribute__((ext_vector_type(16))) _Float16 v16h;
typedef __attribute__((ext_vector_type(8)))  float    v8f;
typedef __attribute__((ext_vector_type(4)))  _Float16 h4;

#define NFD 64
#define EFD 32
#define NHD 64
#define CSD 64

// K-index inside a 32-wide chunk for element i (0..15) of an A/B fragment,
// per ISA 7.12.2 "16-bit A-Matrix 16x32": v0..3 hold K=0..7 (+8 for upper
// half-wave), v4..7 hold K=16..23 (+8 for upper half-wave).
__device__ __forceinline__ int frag_k(int i, int half) {
  int v = i >> 1, j = i & 1;
  int k = (v < 4) ? (v * 2 + j) : (16 + (v - 4) * 2 + j);
  return k + half * 8;
}

// A fragment from row-major LDS tile [16 x stride]: per-lane contiguous k
// ranges -> compiler merges into 2x ds_load_b128.
__device__ __forceinline__ v16h frag_a(const _Float16* sm, int stride, int m,
                                       int kbase, int half) {
  v16h a;
#pragma unroll
  for (int i = 0; i < 16; ++i) a[i] = sm[m * stride + kbase + frag_k(i, half)];
  return a;
}

// ---- B stored pre-swizzled in fragment order ----
// Group of 40 halves (80B) per (kchunk, ncol): half0 frag at +0, half1 frag at
// +16, 8 halves pad. 80B lane stride => bank stride 20 => conflict-free.
__device__ __forceinline__ int bsw_base(int ncols, int kc, int ncol, int half) {
  return (kc * ncols + ncol) * 40 + half * 16;
}
__device__ __forceinline__ v16h frag_b_sw(const _Float16* sm, int ncols, int kc,
                                          int ncol, int half) {
  const _Float16* p = sm + bsw_base(ncols, kc, ncol, half);
  v16h b;
#pragma unroll
  for (int i = 0; i < 16; ++i) b[i] = p[i];  // 32 contiguous bytes
  return b;
}
// Stage weight matrix into swizzled-f16 B layout.
// transposed=false: W is (K x ncols) row-major, B[k][n]=W[k*ncols+n]
// transposed=true : W is (ncols x K) row-major, B[k][n]=W[n*K+k]
__device__ __forceinline__ void stage_b_sw(_Float16* sm, const float* W, int K,
                                           int ncols, int tid, int nthreads,
                                           bool transposed) {
  int total = (K >> 5) * ncols * 32;
  for (int flat = tid; flat < total; flat += nthreads) {
    int kc = flat / (ncols * 32);
    int rem = flat - kc * (ncols * 32);
    int ncol = rem >> 5, t = rem & 31, half = t >> 4, i = t & 15;
    int k = kc * 32 + frag_k(i, half);
    float v = transposed ? W[ncol * K + k] : W[k * ncols + ncol];
    sm[bsw_base(ncols, kc, ncol, half) + i] = (_Float16)v;
  }
}

__device__ __forceinline__ float leaky(float x) { return x > 0.f ? x : 0.01f * x; }
// order-preserving float<->uint mapping for atomic max
__device__ __forceinline__ unsigned f2o(float f) {
  unsigned u = __float_as_uint(f);
  return (u & 0x80000000u) ? ~u : (u | 0x80000000u);
}
__device__ __forceinline__ float o2f(unsigned u) {
  return (u & 0x80000000u) ? __uint_as_float(u & 0x7fffffffu)
                           : __uint_as_float(~u);
}

// ---- (1) h_v = leaky(nf @ W_n + b_n) -> hv16 (f16); also emit nf16 copy ----
__global__ void k_node_embed(const float* __restrict__ nf,
                             const float* __restrict__ Wn,
                             const float* __restrict__ bn,
                             _Float16* __restrict__ hv16,
                             _Float16* __restrict__ nf16, int N) {
  __shared__ _Float16 sW[2 * 64 * 40];
  __shared__ float sb[64];
  __shared__ _Float16 sA[4][16 * 64];
  int tid = threadIdx.x;
  stage_b_sw(sW, Wn, 64, 64, tid, 128, false);
  if (tid < 64) sb[tid] = bn[tid];
  int wv = tid >> 5, lane = tid & 31;
  int row0 = blockIdx.x * 64 + wv * 16;
  const float4* nf4 = (const float4*)nf;
  for (int q = lane; q < 16 * 16; q += 32) {
    int rr = q >> 4, qc = q & 15;
    int r = row0 + rr;
    float4 v = {0.f, 0.f, 0.f, 0.f};
    if (r < N) v = nf4[(size_t)r * 16 + qc];
    h4 hx = {(_Float16)v.x, (_Float16)v.y, (_Float16)v.z, (_Float16)v.w};
    *(h4*)&sA[wv][rr * 64 + qc * 4] = hx;
    if (r < N) *(h4*)&nf16[(size_t)r * 64 + qc * 4] = hx;  // f16 master copy
  }
  __syncthreads();
  int m = lane & 15, half = lane >> 4;
  v16h a0 = frag_a(sA[wv], 64, m, 0, half);
  v16h a1 = frag_a(sA[wv], 64, m, 32, half);
#pragma unroll
  for (int t = 0; t < 4; ++t) {
    int ncol = t * 16 + m;
    float bb = sb[ncol];
    v8f acc = {bb, bb, bb, bb, bb, bb, bb, bb};
    acc = __builtin_amdgcn_wmma_f32_16x16x32_f16(false, a0, false,
          frag_b_sw(sW, 64, 0, ncol, half), (short)0, acc, false, false);
    acc = __builtin_amdgcn_wmma_f32_16x16x32_f16(false, a1, false,
          frag_b_sw(sW, 64, 1, ncol, half), (short)0, acc, false, false);
#pragma unroll
    for (int r = 0; r < 8; ++r)   // transpose via LDS (sA reusable: frags hoisted)
      sA[wv][(r + half * 8) * 64 + ncol] = (_Float16)leaky(acc[r]);
  }
  // wave-internal DS ops are in-order: safe to read back, coalesced 8B stores
  for (int q = lane; q < 16 * 16; q += 32) {
    int rr = q >> 4, qc = q & 15;
    int r = row0 + rr;
    if (r < N)
      *(h4*)&hv16[(size_t)r * 64 + qc * 4] = *(const h4*)&sA[wv][rr * 64 + qc * 4];
  }
}

// ---- (2,3) per edge: h_e, m (f16, stored), logit (stored) + atomic max ----
__global__ void k_edge_pass1(const _Float16* __restrict__ nf16,
                             const float* __restrict__ ef,
                             const int* __restrict__ src,
                             const int* __restrict__ dst,
                             const _Float16* __restrict__ hv16,
                             const float* __restrict__ We,
                             const float* __restrict__ be,
                             const float* __restrict__ Wm,
                             const float* __restrict__ bm,
                             const float* __restrict__ Wl,
                             const float* __restrict__ bl,
                             _Float16* __restrict__ gm16,
                             float* __restrict__ glogit,
                             unsigned* __restrict__ nmax, int E) {
  __shared__ _Float16 sWe[3 * 64 * 40];  // 7680 h
  __shared__ _Float16 sWm[2 * 64 * 40];  // 5120 h
  __shared__ float sWl[128];
  __shared__ float sbe[64], sbm[64];
  __shared__ _Float16 sAe[4][16 * 96];
  __shared__ _Float16 sHe[4][16 * 64];
  __shared__ _Float16 sHv[4][16 * 64];
  int tid = threadIdx.x;
  stage_b_sw(sWe, We, 96, 64, tid, 128, false);
  stage_b_sw(sWm, Wm, 64, 64, tid, 128, false);
  if (tid < 128) sWl[tid] = Wl[tid];
  if (tid < 64) { sbe[tid] = be[tid]; sbm[tid] = bm[tid]; }
  int wv = tid >> 5, lane = tid & 31;
  int e0 = blockIdx.x * 64 + wv * 16;
  if (tid == 0) {  // prefetch next block's index arrays (global_prefetch_b8)
    int pe = blockIdx.x * 64 + 64;
    if (pe < E) { __builtin_prefetch(src + pe, 0, 1); __builtin_prefetch(dst + pe, 0, 1); }
  }
  // gather f16 source-node rows into sAe[:, 0:64]
  for (int q = lane; q < 16 * 16; q += 32) {
    int rr = q >> 4, qc = q & 15;
    int e = e0 + rr; if (e >= E) e = E - 1;
    int s = src[e];
    *(h4*)&sAe[wv][rr * 96 + qc * 4] = *(const h4*)&nf16[(size_t)s * 64 + qc * 4];
  }
  // edge feats (fp32 -> f16) into sAe[:, 64:96]
  const float4* ef4 = (const float4*)ef;
  for (int q = lane; q < 16 * 8; q += 32) {
    int rr = q >> 3, qq = q & 7;
    int e = e0 + rr; if (e >= E) e = E - 1;
    float4 v = ef4[(size_t)e * 8 + qq];
    h4 hx = {(_Float16)v.x, (_Float16)v.y, (_Float16)v.z, (_Float16)v.w};
    *(h4*)&sAe[wv][rr * 96 + 64 + qq * 4] = hx;
  }
  // gather f16 hv[src] rows
  for (int q = lane; q < 16 * 16; q += 32) {
    int rr = q >> 4, qc = q & 15;
    int e = e0 + rr; if (e >= E) e = E - 1;
    int s = src[e];
    *(h4*)&sHv[wv][rr * 64 + qc * 4] = *(const h4*)&hv16[(size_t)s * 64 + qc * 4];
  }
  __syncthreads();
  int m = lane & 15, half = lane >> 4;
  v16h a0 = frag_a(sAe[wv], 96, m, 0, half);
  v16h a1 = frag_a(sAe[wv], 96, m, 32, half);
  v16h a2 = frag_a(sAe[wv], 96, m, 64, half);
  // h_e = leaky(A(16x96) @ We + be) -> sHe (f16)
#pragma unroll
  for (int t = 0; t < 4; ++t) {
    int ncol = t * 16 + m;
    float bb = sbe[ncol];
    v8f acc = {bb, bb, bb, bb, bb, bb, bb, bb};
    acc = __builtin_amdgcn_wmma_f32_16x16x32_f16(false, a0, false,
          frag_b_sw(sWe, 64, 0, ncol, half), (short)0, acc, false, false);
    acc = __builtin_amdgcn_wmma_f32_16x16x32_f16(false, a1, false,
          frag_b_sw(sWe, 64, 1, ncol, half), (short)0, acc, false, false);
    acc = __builtin_amdgcn_wmma_f32_16x16x32_f16(false, a2, false,
          frag_b_sw(sWe, 64, 2, ncol, half), (short)0, acc, false, false);
#pragma unroll
    for (int r = 0; r < 8; ++r)
      sHe[wv][(r + half * 8) * 64 + ncol] = (_Float16)leaky(acc[r]);
  }
  __syncthreads();
  // m = h_e @ Wm + bm -> sAe reused as f16 out-tile, then coalesced store
  v16h h0 = frag_a(sHe[wv], 64, m, 0, half);
  v16h h1 = frag_a(sHe[wv], 64, m, 32, half);
#pragma unroll
  for (int t = 0; t < 4; ++t) {
    int ncol = t * 16 + m;
    float bb = sbm[ncol];
    v8f acc = {bb, bb, bb, bb, bb, bb, bb, bb};
    acc = __builtin_amdgcn_wmma_f32_16x16x32_f16(false, h0, false,
          frag_b_sw(sWm, 64, 0, ncol, half), (short)0, acc, false, false);
    acc = __builtin_amdgcn_wmma_f32_16x16x32_f16(false, h1, false,
          frag_b_sw(sWm, 64, 1, ncol, half), (short)0, acc, false, false);
#pragma unroll
    for (int r = 0; r < 8; ++r)
      sAe[wv][(r + half * 8) * 64 + ncol] = (_Float16)acc[r];
  }
  for (int q = lane; q < 16 * 16; q += 32) {
    int rr = q >> 4, qc = q & 15;
    int e = e0 + rr;
    if (e < E)
      *(h4*)&gm16[(size_t)e * 64 + qc * 4] = *(const h4*)&sAe[wv][rr * 64 + qc * 4];
  }
  // logit = leaky([hv[src] | h_e] @ Wl + bl): one edge per lane 0..15
  if (lane < 16) {
    int e = e0 + lane;
    if (e < E) {
      float dotv = bl[0];
#pragma unroll 8
      for (int c = 0; c < 64; ++c) dotv += (float)sHv[wv][lane * 64 + c] * sWl[c];
#pragma unroll 8
      for (int c = 0; c < 64; ++c) dotv += (float)sHe[wv][lane * 64 + c] * sWl[64 + c];
      float l = leaky(dotv);
      glogit[e] = l;
      atomicMax(&nmax[dst[e]], f2o(l));
    }
  }
}

// ---------------- (4a) ex = exp(logit - max[dst]); denom += ex ----------------
__global__ void k_exp(float* __restrict__ gex, const int* __restrict__ dst,
                      const unsigned* __restrict__ nmax, float* __restrict__ den,
                      int E) {
  int e = blockIdx.x * blockDim.x + threadIdx.x;
  if (e >= E) return;
  int d = dst[e];
  float ex = __expf(gex[e] - o2f(nmax[d]));
  gex[e] = ex;
  atomicAdd(&den[d], ex);
}

// ---------------- (4b,5) C[dst] += (ex/denom[dst]) * m ----------------
__global__ void k_scatter(const float* __restrict__ gex,
                          const float* __restrict__ den,
                          const int* __restrict__ dst,
                          const _Float16* __restrict__ gm16,
                          float* __restrict__ Cb, int E) {
  long long idx = (long long)blockIdx.x * blockDim.x + threadIdx.x;
  if (idx >= (long long)E * 16) return;
  int e = (int)(idx >> 4), q = (int)(idx & 15);
  int d = dst[e];
  float wgt = gex[e] / den[d];
  h4 mm = *(const h4*)&gm16[(size_t)e * 64 + q * 4];
  float* cp = Cb + (size_t)d * 64 + q * 4;
  atomicAdd(cp + 0, wgt * (float)mm[0]);
  atomicAdd(cp + 1, wgt * (float)mm[1]);
  atomicAdd(cp + 2, wgt * (float)mm[2]);
  atomicAdd(cp + 3, wgt * (float)mm[3]);
}

// -------- (6) out = act(A) @ W^T + bias   (W is (192,64) row-major) --------
// apply_elu=1: A = Af32 with elu; apply_elu=0: A = A16 (f16) as-is.
__global__ void k_gru_gemm(const float* __restrict__ Af32,
                           const _Float16* __restrict__ A16,
                           const float* __restrict__ W,
                           const float* __restrict__ bias,
                           float* __restrict__ out, int N, int apply_elu) {
  __shared__ _Float16 sB[2 * 192 * 40];  // 15360 h
  __shared__ float sb[192];
  __shared__ _Float16 sA[4][16 * 64];
  int tid = threadIdx.x;
  stage_b_sw(sB, W, 64, 192, tid, 128, true);
  for (int i = tid; i < 192; i += 128) sb[i] = bias[i];
  int wv = tid >> 5, lane = tid & 31;
  int row0 = blockIdx.x * 64 + wv * 16;
  if (apply_elu) {
    const float4* A4 = (const float4*)Af32;
    for (int q = lane; q < 16 * 16; q += 32) {
      int rr = q >> 4, qc = q & 15;
      int r = row0 + rr;
      float4 v = {0.f, 0.f, 0.f, 0.f};
      if (r < N) v = A4[(size_t)r * 16 + qc];
      float x0 = v.x > 0.f ? v.x : expm1f(v.x);
      float x1 = v.y > 0.f ? v.y : expm1f(v.y);
      float x2 = v.z > 0.f ? v.z : expm1f(v.z);
      float x3 = v.w > 0.f ? v.w : expm1f(v.w);
      h4 hx = {(_Float16)x0, (_Float16)x1, (_Float16)x2, (_Float16)x3};
      *(h4*)&sA[wv][rr * 64 + qc * 4] = hx;
    }
  } else {
    for (int q = lane; q < 16 * 16; q += 32) {
      int rr = q >> 4, qc = q & 15;
      int r = row0 + rr;
      h4 hx = {(_Float16)0.f, (_Float16)0.f, (_Float16)0.f, (_Float16)0.f};
      if (r < N) hx = *(const h4*)&A16[(size_t)r * 64 + qc * 4];
      *(h4*)&sA[wv][rr * 64 + qc * 4] = hx;
    }
  }
  __syncthreads();
  int m = lane & 15, half = lane >> 4;
  v16h a0 = frag_a(sA[wv], 64, m, 0, half);
  v16h a1 = frag_a(sA[wv], 64, m, 32, half);
  bool full = (row0 + 16) <= N;  // wave-uniform fast path (no per-row exec masks)
#pragma unroll
  for (int t = 0; t < 12; ++t) {
    int ncol = t * 16 + m;
    float bb = sb[ncol];
    v8f acc = {bb, bb, bb, bb, bb, bb, bb, bb};
    acc = __builtin_amdgcn_wmma_f32_16x16x32_f16(false, a0, false,
          frag_b_sw(sB, 192, 0, ncol, half), (short)0, acc, false, false);
    acc = __builtin_amdgcn_wmma_f32_16x16x32_f16(false, a1, false,
          frag_b_sw(sB, 192, 1, ncol, half), (short)0, acc, false, false);
    if (full) {
#pragma unroll
      for (int r = 0; r < 8; ++r)
        out[(size_t)(row0 + r + half * 8) * 192 + ncol] = acc[r];
    } else {
#pragma unroll
      for (int r = 0; r < 8; ++r) {
        int gr = row0 + r + half * 8;
        if (gr < N) out[(size_t)gr * 192 + ncol] = acc[r];
      }
    }
  }
}

// ---------------- (6) GRU gate math + relu ----------------
__global__ void k_gates(const float* __restrict__ gi, const float* __restrict__ gh,
                        const _Float16* __restrict__ hv16,
                        float* __restrict__ out, int N) {
  int idx = blockIdx.x * blockDim.x + threadIdx.x;
  if (idx >= N * 64) return;
  int n = idx >> 6, j = idx & 63;
  size_t base = (size_t)n * 192;
  float ir = gi[base + j], iz = gi[base + 64 + j], in_ = gi[base + 128 + j];
  float hr = gh[base + j], hz = gh[base + 64 + j], hn = gh[base + 128 + j];
  float r = 1.f / (1.f + __expf(-(ir + hr)));
  float z = 1.f / (1.f + __expf(-(iz + hz)));
  float nn = tanhf(in_ + r * hn);
  float h = (1.f - z) * nn + z * (float)hv16[idx];
  out[idx] = h > 0.f ? h : 0.f;
}

extern "C" void kernel_launch(void* const* d_in, const int* in_sizes, int n_in,
                              void* d_out, int out_size, void* d_ws,
                              size_t ws_size, hipStream_t stream) {
  const float* node_feats = (const float*)d_in[0];
  const float* edge_feats = (const float*)d_in[1];
  const int*   src  = (const int*)d_in[2];
  const int*   dst  = (const int*)d_in[3];
  const float* W_n  = (const float*)d_in[4];
  const float* b_n  = (const float*)d_in[5];
  const float* W_e  = (const float*)d_in[6];
  const float* b_e  = (const float*)d_in[7];
  const float* W_l  = (const float*)d_in[8];
  const float* b_l  = (const float*)d_in[9];
  const float* W_m  = (const float*)d_in[10];
  const float* b_m  = (const float*)d_in[11];
  const float* W_ih = (const float*)d_in[12];
  const float* W_hh = (const float*)d_in[13];
  const float* b_ih = (const float*)d_in[14];
  const float* b_hh = (const float*)d_in[15];
  int N = in_sizes[0] / NFD;
  int E = in_sizes[2];
  float* out = (float*)d_out;

  auto al = [](size_t x) { return (x + 255) & ~(size_t)255; };
  char* w = (char*)d_ws;
  _Float16* hv16 = (_Float16*)w; w += al((size_t)N * NHD * 2);
  _Float16* nf16 = (_Float16*)w; w += al((size_t)N * NFD * 2);
  _Float16* gm16 = (_Float16*)w; w += al((size_t)E * CSD * 2);
  float*    gex  = (float*)w;    w += al((size_t)E * 4);      // logit, then ex
  unsigned* nmax = (unsigned*)w; w += al((size_t)N * 4);
  float*    den  = (float*)w;    w += al((size_t)N * 4);
  float*    Cb   = (float*)w;    w += al((size_t)N * CSD * 4);
  float*    gi   = (float*)w;    w += al((size_t)N * 3 * NHD * 4);
  float*    gh   = (float*)w;    w += al((size_t)N * 3 * NHD * 4);

  hipMemsetAsync(nmax, 0, (size_t)N * 4, stream);  // 0 == ordered -inf identity
  hipMemsetAsync(den, 0, (size_t)N * 4, stream);
  hipMemsetAsync(Cb, 0, (size_t)N * CSD * 4, stream);

  k_node_embed<<<(N + 63) / 64, 128, 0, stream>>>(node_feats, W_n, b_n, hv16,
                                                  nf16, N);
  k_edge_pass1<<<(E + 63) / 64, 128, 0, stream>>>(
      nf16, edge_feats, src, dst, hv16, W_e, b_e, W_m, b_m, W_l, b_l, gm16,
      gex, nmax, E);
  k_exp<<<(E + 255) / 256, 256, 0, stream>>>(gex, dst, nmax, den, E);
  {
    long long tot = (long long)E * 16;
    k_scatter<<<(unsigned)((tot + 255) / 256), 256, 0, stream>>>(gex, den, dst,
                                                                 gm16, Cb, E);
  }
  k_gru_gemm<<<(N + 63) / 64, 128, 0, stream>>>(Cb, hv16, W_ih, b_ih, gi, N, 1);
  k_gru_gemm<<<(N + 63) / 64, 128, 0, stream>>>(Cb, hv16, W_hh, b_hh, gh, N, 0);
  k_gates<<<(N * NHD + 255) / 256, 256, 0, stream>>>(gi, gh, hv16, out, N);
  // second output: passthrough edge features
  hipMemcpyAsync(out + (size_t)N * NHD, edge_feats, (size_t)E * EFD * 4,
                 hipMemcpyDeviceToDevice, stream);
}
